// CutModel_73383811219613
// MI455X (gfx1250) — matrix operand order repaired and checked by reading, not compiled, not run on hardware
//
#include <hip/hip_runtime.h>

// ---------------------------------------------------------------------------
// Problem constants (match reference)
// ---------------------------------------------------------------------------
#define NNODES  50000
#define NEDGES  1600000
#define HDIM    128
#define KSEL    5000
#define DELTA_C 2.0f

typedef _Float16 v16h __attribute__((ext_vector_type(16)));
typedef float    v8f  __attribute__((ext_vector_type(8)));
typedef float    v4f  __attribute__((ext_vector_type(4)));

static __device__ __forceinline__ unsigned f32_ord(float f) {
  unsigned u = __float_as_uint(f);
  return (u & 0x80000000u) ? ~u : (u | 0x80000000u);
}

// ---------------------------------------------------------------------------
// Pack a 128x128 row-major f32 weight (W[k*128+n]) into per-wave WMMA B
// fragments:  P[((s*8+t)*32+lane)*16+e]
//   lane 0-15 : column n = t*16+lane,    K = s*32 + (e&7) + (e>>3)*16      (e = 0..15)
//   lane16-31 : column n = t*16+lane-16, K = s*32 + 8 + (e&7) + (e>>3)*16
// ---------------------------------------------------------------------------
__global__ void pack_w_kernel(const float* __restrict__ W, _Float16* __restrict__ P)
{
  int i = blockIdx.x * blockDim.x + threadIdx.x;
  if (i >= 8 * 4 * 32 * 16) return;
  int e    = i & 15;
  int lane = (i >> 4) & 31;
  int t    = (i >> 9) & 7;
  int s    = (i >> 12) & 3;
  int n = t * 16 + (lane & 15);
  int k = s * 32 + (lane >> 4) * 8 + (e & 7) + (e >> 3) * 16;
  P[i] = (_Float16)W[k * HDIM + n];
}

// ---------------------------------------------------------------------------
// Degree / inverse degree
// ---------------------------------------------------------------------------
__global__ void degree_kernel(const int* __restrict__ dst, float* __restrict__ deg, int ne)
{
  int e = blockIdx.x * blockDim.x + threadIdx.x;
  if (e < ne) atomicAdd(&deg[dst[e]], 1.0f);
}

__global__ void invdeg_kernel(const float* __restrict__ deg, float* __restrict__ inv, int n)
{
  int i = blockIdx.x * blockDim.x + threadIdx.x;
  if (i < n) inv[i] = 1.0f / fmaxf(deg[i], 1.0f);
}

// ---------------------------------------------------------------------------
// Edge scatter:  agg[dst] += X[src]   (one wave32 per edge, 4 floats/lane)
// ---------------------------------------------------------------------------
__global__ __launch_bounds__(256)
void scatter_add_kernel(const float* __restrict__ X, const int* __restrict__ src,
                        const int* __restrict__ dst, float* __restrict__ AGG, int ne)
{
  int e = blockIdx.x * 8 + (threadIdx.x >> 5);
  if (e >= ne) return;
  int lane = threadIdx.x & 31;
  int s = src[e], d = dst[e];
  const v4f v = ((const v4f*)(X + (size_t)s * HDIM))[lane];
  float* out = AGG + (size_t)d * HDIM + lane * 4;
  atomicAdd(out + 0, v[0]);
  atomicAdd(out + 1, v[1]);
  atomicAdd(out + 2, v[2]);
  atomicAdd(out + 3, v[3]);
}

// ---------------------------------------------------------------------------
// Fused GEMM:  Out = act( (A + coef(row)*A2) @ W + bias ),  coef = alpha2*rowscale
// One wave per 16-row M tile, full 128 columns:
//   8 f32 accum tiles x 4 K-steps -> 32 x v_wmma_f32_16x16x32_f16 per wave.
// Branch-free A path: when A2 is null we alias it to A with coef=0 so the
// fragment build is pure b128 loads + FMA + pack-convert (no divergence).
// ---------------------------------------------------------------------------
__global__ __launch_bounds__(256)
void gemm_kernel(const float* __restrict__ A, const float* __restrict__ A2,
                 const float* __restrict__ rowscale, float alpha2,
                 const _Float16* __restrict__ Bpack, const float* __restrict__ bias,
                 float* __restrict__ Out, int mtiles, int do_elu)
{
  int wave = blockIdx.x * 8 + (threadIdx.x >> 5);
  if (wave >= mtiles) return;
  int lane = threadIdx.x & 31;
  int l15  = lane & 15;
  int hi   = lane >> 4;
  int mrow = wave * 16 + l15;

  const float* arow  = A + (size_t)mrow * HDIM;
  const float* a2row = A2 ? (A2 + (size_t)mrow * HDIM) : arow;  // alias when unused
  float coef = A2 ? (alpha2 * (rowscale ? rowscale[mrow] : 1.0f)) : 0.0f;

  v8f acc[8];
#pragma unroll
  for (int t = 0; t < 8; ++t) {
    float bv = bias[t * 16 + l15];
    v8f a0;
#pragma unroll
    for (int e = 0; e < 8; ++e) a0[e] = bv;
    acc[t] = a0;
  }

  const v16h* Bp = (const v16h*)Bpack;
#pragma unroll
  for (int s = 0; s < 4; ++s) {
    // A fragment (ISA 16-bit A 16x32 layout): elements 0..7 -> K=k0..k0+7,
    // elements 8..15 -> K=k0+16..k0+23, k0 = s*32 + hi*8
    int k0 = s * 32 + hi * 8;
    v4f p0 = *(const v4f*)(arow  + k0);
    v4f p1 = *(const v4f*)(arow  + k0 + 4);
    v4f p2 = *(const v4f*)(arow  + k0 + 16);
    v4f p3 = *(const v4f*)(arow  + k0 + 20);
    v4f q0 = *(const v4f*)(a2row + k0);
    v4f q1 = *(const v4f*)(a2row + k0 + 4);
    v4f q2 = *(const v4f*)(a2row + k0 + 16);
    v4f q3 = *(const v4f*)(a2row + k0 + 20);
    v16h afrag;
#pragma unroll
    for (int e = 0; e < 4; ++e) {
      afrag[e]      = (_Float16)(p0[e] + coef * q0[e]);
      afrag[4 + e]  = (_Float16)(p1[e] + coef * q1[e]);
      afrag[8 + e]  = (_Float16)(p2[e] + coef * q2[e]);
      afrag[12 + e] = (_Float16)(p3[e] + coef * q3[e]);
    }
#pragma unroll
    for (int t = 0; t < 8; ++t) {
      v16h bfrag = Bp[(s * 8 + t) * 32 + lane];
      acc[t] = __builtin_amdgcn_wmma_f32_16x16x32_f16(
          false, afrag, false, bfrag, (short)0, acc[t], false, false);
    }
  }

  // Epilogue: C/D layout -> VGPR e holds row (hi*8+e), column l15 of tile t
  int rbase = wave * 16 + hi * 8;
#pragma unroll
  for (int t = 0; t < 8; ++t) {
    int col = t * 16 + l15;
#pragma unroll
    for (int e = 0; e < 8; ++e) {
      float v = acc[t][e];
      if (do_elu) v = (v > 0.0f) ? v : (__expf(v) - 1.0f);
      Out[(size_t)(rbase + e) * HDIM + col] = v;
    }
  }
}

// ---------------------------------------------------------------------------
// score = tanh(H @ sc_w + sc_b)  (one wave per node; 32-lane shuffle reduce)
// Also emit order-preserving u32 keys for top-k.
// ---------------------------------------------------------------------------
__global__ __launch_bounds__(256)
void score_kernel(const float* __restrict__ Hh, const float* __restrict__ scw,
                  const float* __restrict__ scb, float* __restrict__ score_out,
                  unsigned* __restrict__ ord_out, int n)
{
  int node = blockIdx.x * 8 + (threadIdx.x >> 5);
  if (node >= n) return;
  int lane = threadIdx.x & 31;
  const v4f hv = ((const v4f*)(Hh + (size_t)node * HDIM))[lane];
  const v4f wv = ((const v4f*)scw)[lane];
  float p = hv[0] * wv[0] + hv[1] * wv[1] + hv[2] * wv[2] + hv[3] * wv[3];
#pragma unroll
  for (int off = 16; off > 0; off >>= 1) p += __shfl_xor(p, off, 32);
  if (lane == 0) {
    float sc = tanhf(p + scb[0]);
    score_out[node] = sc;
    ord_out[node]   = f32_ord(sc);
  }
}

// ---------------------------------------------------------------------------
// aux_loss partial: sum over edges of s[src]*s[dst]
// ---------------------------------------------------------------------------
__global__ void aux_kernel(const float* __restrict__ s, const int* __restrict__ src,
                           const int* __restrict__ dst, float* __restrict__ acc, int ne)
{
  __shared__ float red[256];
  int e = blockIdx.x * 256 + threadIdx.x;
  float v = 0.0f;
  if (e < ne) v = s[src[e]] * s[dst[e]];
  red[threadIdx.x] = v;
  __syncthreads();
  for (int st = 128; st > 0; st >>= 1) {
    if (threadIdx.x < st) red[threadIdx.x] += red[threadIdx.x + st];
    __syncthreads();
  }
  if (threadIdx.x == 0) atomicAdd(acc, red[0]);
}

__global__ void aux_final_kernel(const float* __restrict__ acc, float* __restrict__ out, int ne)
{
  if (threadIdx.x == 0 && blockIdx.x == 0) out[0] = acc[0] / (float)ne;
}

// ---------------------------------------------------------------------------
// Exact top-K (K=5000 of N=50000), single workgroup of 1024 threads.
// Composite key: (ord(score) << 17) | (N-1-i)  -> distinct; JAX tie-break
// (value desc, index asc) reproduced. 49-bit greedy binary search for the
// K-th largest key, then bitonic sort of the K winners (padded to 8192) in LDS.
// ---------------------------------------------------------------------------
#define TOPK_T 1024
__global__ __launch_bounds__(TOPK_T)
void topk_kernel(const unsigned* __restrict__ ord, float* __restrict__ out_kept,
                 int* __restrict__ kept_idx)
{
  __shared__ unsigned long long keys[8192];   // 64KB (WGP LDS = 320KB)
  __shared__ unsigned cnt;
  __shared__ unsigned sel;
  const int tid = threadIdx.x;

  unsigned long long cur = 0ull;
  for (int b = 48; b >= 0; --b) {
    unsigned long long cand = cur | (1ull << b);
    if (tid == 0) cnt = 0;
    __syncthreads();
    unsigned local = 0;
    for (int i = tid; i < NNODES; i += TOPK_T) {
      unsigned long long ck =
          ((unsigned long long)ord[i] << 17) | (unsigned)(NNODES - 1 - i);
      if (ck >= cand) ++local;
    }
    atomicAdd(&cnt, local);
    __syncthreads();
    if (cnt >= (unsigned)KSEL) cur = cand;   // uniform across threads
    __syncthreads();
  }

  if (tid == 0) sel = 0;
  __syncthreads();
  for (int i = tid; i < NNODES; i += TOPK_T) {
    unsigned long long ck =
        ((unsigned long long)ord[i] << 17) | (unsigned)(NNODES - 1 - i);
    if (ck >= cur) {
      unsigned p = atomicAdd(&sel, 1u);
      keys[p] = ck;                          // exactly KSEL selected (keys distinct)
    }
  }
  __syncthreads();
  for (int i = tid; i < 8192; i += TOPK_T)
    if (i >= (int)sel) keys[i] = 0ull;
  __syncthreads();

  // Bitonic sort, descending
  for (int k = 2; k <= 8192; k <<= 1) {
    for (int j = k >> 1; j > 0; j >>= 1) {
      for (int i = tid; i < 8192; i += TOPK_T) {
        int ixj = i ^ j;
        if (ixj > i) {
          unsigned long long a = keys[i], b2 = keys[ixj];
          bool desc = ((i & k) == 0);
          if (desc ? (a < b2) : (a > b2)) { keys[i] = b2; keys[ixj] = a; }
        }
      }
      __syncthreads();
    }
  }

  for (int i = tid; i < KSEL; i += TOPK_T) {
    unsigned node = (unsigned)(NNODES - 1) - (unsigned)(keys[i] & 0x1FFFFull);
    out_kept[i] = (float)node;
    kept_idx[i] = (int)node;
  }
}

// ---------------------------------------------------------------------------
// x_pool = X2[kept] * score[kept]
// ---------------------------------------------------------------------------
__global__ void pool_kernel(const float* __restrict__ X2, const float* __restrict__ score,
                            const int* __restrict__ kept, float* __restrict__ out)
{
  int i = blockIdx.x * blockDim.x + threadIdx.x;
  if (i >= KSEL * HDIM) return;
  int j = i >> 7, f = i & 127;
  int node = kept[j];
  out[i] = X2[(size_t)node * HDIM + f] * score[node];
}

// ---------------------------------------------------------------------------
// Host-side orchestration
// ---------------------------------------------------------------------------
extern "C" void kernel_launch(void* const* d_in, const int* in_sizes, int n_in,
                              void* d_out, int out_size, void* d_ws, size_t ws_size,
                              hipStream_t stream)
{
  (void)in_sizes; (void)n_in; (void)out_size; (void)ws_size;

  const float* x    = (const float*)d_in[0];
  const int*   eidx = (const int*)d_in[1];
  const int*   src  = eidx;
  const int*   dst  = eidx + NEDGES;
  // d_in[2] = batch (unused, single graph)
  const float* g1w1 = (const float*)d_in[3];
  const float* g1b1 = (const float*)d_in[4];
  const float* g1w2 = (const float*)d_in[5];
  const float* g1b2 = (const float*)d_in[6];
  const float* g2w1 = (const float*)d_in[7];
  const float* g2b1 = (const float*)d_in[8];
  const float* g2w2 = (const float*)d_in[9];
  const float* g2b2 = (const float*)d_in[10];
  const float* s1w  = (const float*)d_in[11];
  const float* s1b  = (const float*)d_in[12];
  const float* s2w  = (const float*)d_in[13];
  const float* s2b  = (const float*)d_in[14];
  const float* scw  = (const float*)d_in[15];
  const float* scb  = (const float*)d_in[16];

  // Workspace carve-up (deterministic)
  char* ws = (char*)d_ws;
  size_t off = 0;
  auto take = [&](size_t bytes) {
    size_t o = off;
    off = (off + bytes + 511) & ~(size_t)511;
    return o;
  };
  float*    deg  = (float*)(ws + take((size_t)NNODES * 4));
  float*    inv  = (float*)(ws + take((size_t)NNODES * 4));
  float*    auxa = (float*)(ws + take(64));
  unsigned* ord  = (unsigned*)(ws + take((size_t)NNODES * 4));
  int*      kept = (int*)(ws + take((size_t)KSEL * 4));
  _Float16* pw   = (_Float16*)(ws + take((size_t)6 * 16384 * 2));
  float*    agg  = (float*)(ws + take((size_t)NNODES * HDIM * 4));
  float*    bufA = (float*)(ws + take((size_t)NNODES * HDIM * 4));
  float*    bufB = (float*)(ws + take((size_t)NNODES * HDIM * 4));
  float*    bufC = (float*)(ws + take((size_t)NNODES * HDIM * 4));

  float* out       = (float*)d_out;
  float* out_pool  = out;                             // [KSEL*HDIM]
  float* out_kept  = out_pool + (size_t)KSEL * HDIM;  // [KSEL]
  float* out_score = out_kept + KSEL;                 // [NNODES]
  float* out_aux   = out_score + NNODES;              // [1]

  const size_t aggBytes = (size_t)NNODES * HDIM * 4;
  const int MT = NNODES / 16;                         // 3125 M tiles, exact
  dim3 blk(256);
  dim3 gGemm((MT + 7) / 8);
  dim3 gScat((NEDGES + 7) / 8);

  // Pack weights into WMMA fragment order (tiny, once per launch)
  const float* Wsrc[6] = {g1w1, g1w2, g2w1, g2w2, s1w, s2w};
  for (int m = 0; m < 6; ++m)
    pack_w_kernel<<<64, 256, 0, stream>>>(Wsrc[m], pw + (size_t)m * 16384);

  // Degrees
  hipMemsetAsync(deg, 0, (size_t)NNODES * 4, stream);
  hipMemsetAsync(auxa, 0, 4, stream);
  degree_kernel<<<(NEDGES + 255) / 256, 256, 0, stream>>>(dst, deg, NEDGES);
  invdeg_kernel<<<(NNODES + 255) / 256, 256, 0, stream>>>(deg, inv, NNODES);

  // GIN layer 1:  x1 = elu( elu((x+agg)@w1+b1) @ w2 + b2 )
  hipMemsetAsync(agg, 0, aggBytes, stream);
  scatter_add_kernel<<<gScat, blk, 0, stream>>>(x, src, dst, agg, NEDGES);
  gemm_kernel<<<gGemm, blk, 0, stream>>>(x, agg, nullptr, 1.0f,
                                         pw + 0 * 16384, g1b1, bufA, MT, 1);
  gemm_kernel<<<gGemm, blk, 0, stream>>>(bufA, nullptr, nullptr, 0.0f,
                                         pw + 1 * 16384, g1b2, bufB, MT, 1);

  // GIN layer 2:  x2 = elu( elu((x1+agg)@w1+b1) @ w2 + b2 )   (x2 kept for pooling)
  hipMemsetAsync(agg, 0, aggBytes, stream);
  scatter_add_kernel<<<gScat, blk, 0, stream>>>(bufB, src, dst, agg, NEDGES);
  gemm_kernel<<<gGemm, blk, 0, stream>>>(bufB, agg, nullptr, 1.0f,
                                         pw + 2 * 16384, g2b1, bufA, MT, 1);
  gemm_kernel<<<gGemm, blk, 0, stream>>>(bufA, nullptr, nullptr, 0.0f,
                                         pw + 3 * 16384, g2b2, bufC, MT, 1);

  // Het layer 1:  h1 = elu( (x2 - DELTA*inv_deg*agg) @ s1_w + s1_b )
  hipMemsetAsync(agg, 0, aggBytes, stream);
  scatter_add_kernel<<<gScat, blk, 0, stream>>>(bufC, src, dst, agg, NEDGES);
  gemm_kernel<<<gGemm, blk, 0, stream>>>(bufC, agg, inv, -DELTA_C,
                                         pw + 4 * 16384, s1b, bufA, MT, 1);

  // Het layer 2:  h2 = elu( (h1 - DELTA*inv_deg*agg) @ s2_w + s2_b )
  hipMemsetAsync(agg, 0, aggBytes, stream);
  scatter_add_kernel<<<gScat, blk, 0, stream>>>(bufA, src, dst, agg, NEDGES);
  gemm_kernel<<<gGemm, blk, 0, stream>>>(bufA, agg, inv, -DELTA_C,
                                         pw + 5 * 16384, s2b, bufB, MT, 1);

  // Score, aux loss, exact top-K, pooling
  score_kernel<<<(NNODES + 7) / 8, 256, 0, stream>>>(bufB, scw, scb, out_score, ord, NNODES);
  aux_kernel<<<(NEDGES + 255) / 256, 256, 0, stream>>>(out_score, src, dst, auxa, NEDGES);
  aux_final_kernel<<<1, 32, 0, stream>>>(auxa, out_aux, NEDGES);
  topk_kernel<<<1, TOPK_T, 0, stream>>>(ord, out_kept, kept);
  pool_kernel<<<(KSEL * HDIM + 255) / 256, 256, 0, stream>>>(bufC, out_score, kept, out_pool);
}
